// KANConv2d_257698038307
// MI455X (gfx1250) — matrix operand order repaired and checked by reading, not compiled, not run on hardware
//
#include <hip/hip_runtime.h>

typedef __attribute__((ext_vector_type(16))) _Float16 v16h;
typedef __attribute__((ext_vector_type(8)))  _Float16 v8h;
typedef __attribute__((ext_vector_type(8)))  float    v8f;

#define NB      8          // spline bases per scalar
#define K2      9          // 3x3 patch
#define CIN     16
#define COUT    16
#define KTOT    1296       // CIN * K2 * (1 + NB)
#define NCHUNK  41         // ceil(KTOT / 32)
#define FSTRIDE 1336       // F row stride (f16): 16B-aligned, 668 dwords -> 16 rows hit distinct banks
#define WELEMS  (NCHUNK * 512)  // packed weight f16 count: [41][2][16][16]
#define PIXPB   64         // pixels per block
#define THREADS 256

// ---------------------------------------------------------------------------
// Kernel 1: fold weights over the KANLinear output dim m and pre-swizzle into
// WMMA B-fragment order: element (k = 32c + 16*half + kk, n = col) lives at
// wpack[((c*2 + half)*16 + col)*16 + kk].  K order: k = (j*9 + t)*9 + s,
// s==0 -> SiLU weight (base), s=1..8 -> spline weight. k >= 1296 -> 0 pad.
// ---------------------------------------------------------------------------
__global__ void kan_pack_weights(const float* __restrict__ base_weight,
                                 const float* __restrict__ spline_weight,
                                 const float* __restrict__ spline_scaler,
                                 _Float16* __restrict__ wpack)
{
    int idx = blockIdx.x * blockDim.x + threadIdx.x;
    if (idx >= WELEMS) return;
    int kk = idx & 15;
    int col = (idx >> 4) & 15;      // n = Cout index
    int hf  = (idx >> 8) & 1;       // K half within 32-chunk
    int c   = idx >> 9;             // chunk
    int k   = c * 32 + hf * 16 + kk;
    float v = 0.0f;
    if (k < KTOT) {
        int pair = k / K2; int s = k - pair * K2;
        int j = pair / K2; int t = pair - j * K2;
        int base = (col * CIN + j) * (K2 * K2) + t;    // [i][j][m=0][t]
        if (s == 0) {
#pragma unroll
            for (int m = 0; m < K2; ++m) v += base_weight[base + m * K2];
        } else {
#pragma unroll
            for (int m = 0; m < K2; ++m)
                v += spline_weight[(base + m * K2) * NB + (s - 1)] *
                     spline_scaler[base + m * K2];
        }
    }
    wpack[idx] = (_Float16)v;
}

// ---------------------------------------------------------------------------
// Kernel 2: per block of 64 pixels:
//  - async-copy packed weights global->LDS (ASYNCcnt path)
//  - zero feature tile with vectorized LDS stores
//  - phase 1 (8 waves): SiLU + 4-nonzero-basis uniform-B-spline scatter (f16)
//  - phase 2 (waves 0..3): fully unrolled 41x v_wmma_f32_16x16x32_f16
// ---------------------------------------------------------------------------
__global__ void __launch_bounds__(THREADS)
kan_conv_wmma(const float* __restrict__ x, const _Float16* __restrict__ wpack,
              const float* __restrict__ bias, float* __restrict__ out)
{
    extern __shared__ _Float16 smem[];
    _Float16* Wl = smem;                 // WELEMS f16 (41984 B)
    _Float16* F  = smem + WELEMS;        // PIXPB * FSTRIDE f16

    const int tid = threadIdx.x;
    const int blk = blockIdx.x;

    // --- async stage-in of packed weights into LDS (16B / lane / op) ---
    {
        const unsigned ldsBase = (unsigned)(uintptr_t)Wl;   // addr[31:0] = LDS offset
        const uintptr_t gBase  = (uintptr_t)wpack;
        const int n16 = WELEMS * 2 / 16;                    // 2624 16B chunks
        for (int i = tid; i < n16; i += THREADS) {
            unsigned lds = ldsBase + (unsigned)(i * 16);
            unsigned long long ga = (unsigned long long)(gBase + (uintptr_t)i * 16);
            asm volatile("global_load_async_to_lds_b128 %0, %1, off"
                         :: "v"(lds), "v"(ga) : "memory");
        }
    }

    // --- zero the feature tile (covers K pad tail + stride padding) ---
    {
        uint4 zz; zz.x = zz.y = zz.z = zz.w = 0u;
        uint4* f4 = (uint4*)F;
        const int n16 = PIXPB * FSTRIDE * 2 / 16;           // 10688
        for (int i = tid; i < n16; i += THREADS) f4[i] = zz;
    }
    asm volatile("s_wait_asynccnt 0" ::: "memory");         // our wave's async W loads done
    __syncthreads();

    // --- phase 1: SiLU + cubic B-spline (only 4 bases non-zero per x) ---
    const int ITEMS = PIXPB * CIN * K2 / THREADS;           // 9216/256 = 36
#pragma unroll 1
    for (int it = 0; it < ITEMS; ++it) {
        int e = tid + it * THREADS;
        int pm = e / (CIN * K2);
        int pair = e - pm * (CIN * K2);
        int j = pair / K2; int t = pair - j * K2;
        int p = blk * PIXPB + pm;
        int b = p >> 14; int h = (p >> 7) & 127; int w = p & 127;
        int dh = t / 3;  int dw = t - dh * 3;               // t = dh*3 + dw (ref order)
        int hs = h + dh - 1, ws = w + dw - 1;               // pad=1
        float xv = 0.0f;
        if ((unsigned)hs < 128u && (unsigned)ws < 128u)
            xv = x[((b * CIN + j) << 14) + (hs << 7) + ws];

        int fo = pm * FSTRIDE + pair * (NB + 1);
        F[fo] = (_Float16)(xv / (1.0f + __expf(-xv)));      // SiLU

        // uniform grid: g[i] = (i-3)*0.4 - 1, knot-unit position from g[0]=-2.2
        float u  = (xv + 2.2f) * 2.5f;
        float fs = floorf(u);
        int seg  = (int)fs;                 // deg-0 interval index (valid 0..10)
        float tt = u - fs;
        float t2 = tt * tt, t3 = t2 * tt;
        float omt = 1.0f - tt;
        const float s6 = 1.0f / 6.0f;
        float p0 = t3 * s6;                                   // basis seg
        float p1 = (-3.f*t3 + 3.f*t2 + 3.f*tt + 1.f) * s6;    // basis seg-1
        float p2 = ( 3.f*t3 - 6.f*t2 + 4.f) * s6;             // basis seg-2
        float p3 = omt * omt * omt * s6;                      // basis seg-3
        if ((unsigned) seg      < 8u) F[fo + 1 + seg] = (_Float16)p0;
        if ((unsigned)(seg - 1) < 8u) F[fo + seg]     = (_Float16)p1;
        if ((unsigned)(seg - 2) < 8u) F[fo + seg - 1] = (_Float16)p2;
        if ((unsigned)(seg - 3) < 8u) F[fo + seg - 2] = (_Float16)p3;
    }
    __syncthreads();

    // --- phase 2: WMMA contraction, waves 0..3 own 16 pixels each ---
    const int wv = tid >> 5;
    if (wv < 4) {
        const int lane = tid & 31;
        const int col  = lane & 15;     // N index for B / C / D
        const int hf   = lane >> 4;     // lane half
        const int m    = lane & 15;     // A row

        v8f acc = {};
        // A order chosen here (contiguous k, same as B): lanes 0-15 hold K 0..15, lanes 16-31 K 16..31
        const _Float16* fptr = F + (wv * 16 + m) * FSTRIDE + hf * 16;
        // B order chosen here (same contiguous k as A): lanes 0-15 hold K 0..15 of col, lanes 16-31 K 16..31
        const _Float16* wptr = Wl + (hf * 16 + col) * 16;
#pragma unroll
        for (int c = 0; c < NCHUNK; ++c) {
            v8h a0 = *(const v8h*)(fptr + c * 32);
            v8h a1 = *(const v8h*)(fptr + c * 32 + 8);
            v16h A = __builtin_shufflevector(a0, a1,
                      0,1,2,3,4,5,6,7,8,9,10,11,12,13,14,15);
            v16h B = *(const v16h*)(wptr + c * 512);
            acc = __builtin_amdgcn_wmma_f32_16x16x32_f16(
                      /*neg_a=*/false, A, /*neg_b=*/false, B,
                      /*c_mod=*/(short)0, acc, /*reuse_a=*/false, /*reuse_b=*/false);
        }

        float bv = bias[col];
        // C/D layout: VGPR r, lanes 0-15 -> M=r, lanes 16-31 -> M=r+8
        int pbase = blk * PIXPB + wv * 16 + (hf << 3);
#pragma unroll
        for (int r = 0; r < 8; ++r) {
            int p = pbase + r;
            int b = p >> 14; int hw = p & 16383;
            out[((b * COUT + col) << 14) + hw] = acc[r] + bv;
        }
    }
}

// ---------------------------------------------------------------------------
extern "C" void kernel_launch(void* const* d_in, const int* in_sizes, int n_in,
                              void* d_out, int out_size, void* d_ws, size_t ws_size,
                              hipStream_t stream)
{
    (void)in_sizes; (void)n_in; (void)out_size; (void)ws_size;
    const float* x        = (const float*)d_in[0];
    const float* base_w   = (const float*)d_in[1];
    const float* spline_w = (const float*)d_in[2];
    const float* scaler   = (const float*)d_in[3];
    const float* bias     = (const float*)d_in[4];
    float* out            = (float*)d_out;
    _Float16* wpack       = (_Float16*)d_ws;    // 41984 bytes of scratch

    kan_pack_weights<<<(WELEMS + 255) / 256, 256, 0, stream>>>(
        base_w, spline_w, scaler, wpack);

    const int npix = 4 * 128 * 128;             // B * oh * ow
    const size_t smem = (size_t)(WELEMS + PIXPB * FSTRIDE) * sizeof(_Float16); // ~208 KB (<320 KB/WG)
    kan_conv_wmma<<<npix / PIXPB, THREADS, smem, stream>>>(x, wpack, bias, out);
}
